// GraphEncoder_44341242364569
// MI455X (gfx1250) — matrix operand order repaired and checked by reading, not compiled, not run on hardware
//
#include <hip/hip_runtime.h>
#include <stdint.h>

// ---------------------------------------------------------------------------
// GraphEncoder for MI455X (gfx1250).
//
// Roofline: FLOPs = 2*N*d^2*Sum_t(13+2K_t) = 2*16384*768^2*72 ~ 1.4 TFLOP.
// Compulsory HBM bytes ~ 0.7 GB -> ~30 us at 23.3 TB/s; bf16 WMMA math is
// sub-microsecond, so the kernel must be HBM-bound, which means minimizing
// *L2* weight re-streaming. This version register-blocks M: 64 rows per
// workgroup as 4x16 sub-tiles, so each B (weight) fragment feeds 4
// v_wmma_f32_16x16x32_bf16 ops -> 4x less L2 weight traffic than a 16-row
// tile and a 4:2 wmma:global-load ratio. Activations stay in LDS (3 x 96 KB
// slots with overlaid live ranges, 314 KB total < 320 KB/WGP); attended and
// rel_mean spill to workspace bf16 (written and re-read by the same
// workgroup -> L2-hot).
// ---------------------------------------------------------------------------

#define D_MODEL   768
#define N_ROWS    16384
#define N_HEADS   12
#define HEAD_DIM  64
#define N_TABLES  4
#define N_EDGES   5
#define TILE_M    64
#define MSUB      4        // 4 sub-tiles of 16 rows
#define NTHREADS  256      // 8 waves (wave32)
#define NWAVES    8

typedef __attribute__((ext_vector_type(16))) __bf16 v16bf;
typedef __attribute__((ext_vector_type(8)))  float  v8f;

struct alignas(16) Frag32 { uint4 lo, hi; };   // 32 bytes = 16 bf16

// ---- bf16 helpers ---------------------------------------------------------
__device__ __forceinline__ unsigned short f2bf(float x) {
    union { float f; unsigned int u; } v; v.f = x;
    unsigned int r = v.u + 0x7FFFu + ((v.u >> 16) & 1u);   // RNE
    return (unsigned short)(r >> 16);
}
__device__ __forceinline__ float bf2f(unsigned short s) {
    union { unsigned int u; float f; } v; v.u = ((unsigned int)s) << 16;
    return v.f;
}

// ---- WMMA fragment loads (ISA 7.12.2 layouts) -----------------------------
// A (16x32 bf16, row-major, stride = aStride):
//   lane half h = lane>>4, row r = lane&15
//   VGPR 0..3 : K = h*8 + {0..7}, VGPR 4..7 : K = 16 + h*8 + {0..7}
__device__ __forceinline__ v16bf load_a(const unsigned short* A, int aStride,
                                        int lane, int kc) {
    const int r = lane & 15, h = lane >> 4;
    const unsigned short* p = A + (size_t)r * aStride + kc + h * 8;
    Frag32 f;
    f.lo = *(const uint4*)(p);        // K = h*8 + 0..7
    f.hi = *(const uint4*)(p + 16);   // K = 16 + h*8 + 0..7
    return __builtin_bit_cast(v16bf, f);
}

// B (32x16 bf16 = W^T slice; W is [outDim][kdim] row-major, global bf16):
//   lane half h, col n = lane&15 -> holds W[n0+n][kc + h*8 + 0..7], W[n0+n][kc + 16 + h*8 + 0..7]
__device__ __forceinline__ v16bf load_b(const unsigned short* W, int kdim,
                                        int n0, int lane, int kc) {
    const int n = lane & 15, h = lane >> 4;
    const unsigned short* p = W + (size_t)(n0 + n) * kdim + kc + h * 8;
    Frag32 f;
    f.lo = *(const uint4*)(p);        // K = h*8 + 0..7
    f.hi = *(const uint4*)(p + 16);   // K = 16 + h*8 + 0..7
    return __builtin_bit_cast(v16bf, f);
}

__device__ __forceinline__ v8f wmma_bf16(v16bf a, v16bf b, v8f c) {
    return __builtin_amdgcn_wmma_f32_16x16x32_bf16(
        false, a, false, b, (short)0, c, false, false);
}

__device__ __forceinline__ void init_acc(v8f acc[MSUB], float bias) {
    #pragma unroll
    for (int s = 0; s < MSUB; ++s)
        #pragma unroll
        for (int v = 0; v < 8; ++v) acc[s][v] = bias;
}

// Register-blocked GEMM: one B fragment feeds MSUB(=4) WMMAs.
// A: [64][aStride] (sub-tile s = rows 16s..16s+15); W: [outDim][wKdim];
// accumulates kLen columns of A against W columns kgBase..kgBase+kLen.
__device__ __forceinline__ void gemm4_acc(const unsigned short* A, int aStride,
                                          int kLen,
                                          const unsigned short* W, int wKdim,
                                          int kgBase, int n0, int lane,
                                          v8f acc[MSUB]) {
    const int n = lane & 15;
    for (int kc = 0; kc < kLen; kc += 32) {
        if (kgBase + kc + 32 < wKdim)   // pull weight stream into near caches
            __builtin_prefetch(W + (size_t)(n0 + n) * wKdim + kgBase + kc + 32, 0, 3);
        v16bf b = load_b(W, wKdim, n0, lane, kgBase + kc);
        #pragma unroll
        for (int s = 0; s < MSUB; ++s) {
            v16bf a = load_a(A + (size_t)s * 16 * aStride, aStride, lane, kc);
            acc[s] = wmma_bf16(a, b, acc[s]);
        }
    }
}

// Store accumulators as bf16 into a [64][stride] row-major buffer.
__device__ __forceinline__ void store_acc_bf16(unsigned short* dst, int stride,
                                               int n0, int lane,
                                               const v8f acc[MSUB], bool relu) {
    const int nlo = lane & 15, hh = lane >> 4;
    #pragma unroll
    for (int s = 0; s < MSUB; ++s)
        #pragma unroll
        for (int v = 0; v < 8; ++v) {
            float x = acc[s][v];
            if (relu) x = fmaxf(x, 0.0f);
            dst[(size_t)(s * 16 + hh * 8 + v) * stride + n0 + nlo] = f2bf(x);
        }
}

// ---------------------------------------------------------------------------
// Prep kernel 1: f32 -> bf16 weight conversion
// ---------------------------------------------------------------------------
__global__ void cvt_bf16_kernel(const float* __restrict__ src,
                                unsigned short* __restrict__ dst, int n) {
    int i = blockIdx.x * blockDim.x + threadIdx.x;
    if (i < n) dst[i] = f2bf(src[i]);
}

// ---------------------------------------------------------------------------
// Prep kernel 2: rw[e][row] = sigmoid(rels[e][row] . rel_w + rel_b)
// ---------------------------------------------------------------------------
__global__ void rw_kernel(const float* __restrict__ rels,
                          const float* __restrict__ rel_w,
                          const float* __restrict__ rel_b,
                          float* __restrict__ rwg) {
    int row  = (blockIdx.x * blockDim.x + threadIdx.x) >> 5;
    int lane = threadIdx.x & 31;
    if (row >= N_EDGES * N_ROWS) return;
    const float* r = rels + (size_t)row * D_MODEL;
    float s = 0.0f;
    for (int c = lane; c < D_MODEL; c += 32) s += r[c] * rel_w[c];
    #pragma unroll
    for (int m = 16; m; m >>= 1) s += __shfl_xor(s, m, 32);
    if (lane == 0) rwg[row] = 1.0f / (1.0f + __expf(-(s + rel_b[0])));
}

// ---------------------------------------------------------------------------
// Main fused kernel: one 64-row tile of one table per workgroup.
// grid = (N_ROWS/64, N_TABLES), block = 256 threads (8 waves)
//
// LDS slot plan (96 KB bf16 [64][768] slots, overlaid live ranges):
//   S1: emb -> kp(per key) -> rebuilt emb (combined blk 0)
//   S2: qp  -> oattn(bf16) -> hidden rows 0..31
//   S3: key/value input    -> hidden rows 32..63   (hid = S2..S3, 64x1536 bf16)
// attended + rel_mean -> workspace bf16 (combined blks 1,2; L2-hot readback)
// ---------------------------------------------------------------------------
#define SLOT_BYTES (TILE_M * D_MODEL * 2u)               // 98304
#define OFF_S1    0u
#define OFF_S2    (OFF_S1 + SLOT_BYTES)
#define OFF_S3    (OFF_S2 + SLOT_BYTES)
#define OFF_SCORE (OFF_S3 + SLOT_BYTES)                  // f32 [3][64][12]
#define OFF_ATTNW (OFF_SCORE + 3u*TILE_M*N_HEADS*4u)     // f32 [3][64][12]
#define OFF_RWV   (OFF_ATTNW + 3u*TILE_M*N_HEADS*4u)     // f32 [3][64]
#define SMEM_BYTES (OFF_RWV + 3u*TILE_M*4u)              // 314112 < 320K

__global__ __launch_bounds__(NTHREADS, 1)
void graph_encoder_main(const float* __restrict__ tables,
                        const float* __restrict__ rels,
                        const unsigned short* __restrict__ Wq,
                        const unsigned short* __restrict__ Wk,
                        const unsigned short* __restrict__ Wv,
                        const unsigned short* __restrict__ Wo,
                        const unsigned short* __restrict__ W1b,
                        const unsigned short* __restrict__ W2b,
                        const unsigned short* __restrict__ Wgb,
                        const float* __restrict__ rwg,
                        unsigned short* __restrict__ att_ws,   // bf16 [4][N][768]
                        unsigned short* __restrict__ rm_ws,    // bf16 [4][N][768]
                        const float* __restrict__ in_b,
                        const float* __restrict__ out_b,
                        const float* __restrict__ b1,
                        const float* __restrict__ b2,
                        const float* __restrict__ gb,
                        float* __restrict__ out) {
    extern __shared__ char smem[];
    unsigned short* S1     = (unsigned short*)(smem + OFF_S1);
    unsigned short* S2     = (unsigned short*)(smem + OFF_S2);
    unsigned short* S3     = (unsigned short*)(smem + OFF_S3);
    unsigned short* hid_bf = (unsigned short*)(smem + OFF_S2); // [64][1536] spans S2+S3
    float*          scores = (float*)(smem + OFF_SCORE);
    float*          attnw  = (float*)(smem + OFF_ATTNW);
    float*          rwv    = (float*)(smem + OFF_RWV);

    const int relK[N_TABLES]        = { 2, 3, 3, 2 };
    const int relOther[N_TABLES][3] = { {1,2,0}, {0,2,3}, {0,1,3}, {1,2,0} };
    const int relEdge [N_TABLES][3] = { {0,1,0}, {0,2,3}, {1,2,4}, {3,4,0} };

    const int t    = blockIdx.y;
    const int row0 = blockIdx.x * TILE_M;
    const int tid  = threadIdx.x;
    const int wave = tid >> 5;
    const int lane = tid & 31;
    const int nlo  = lane & 15;
    const int hh   = lane >> 4;
    const int K    = relK[t];

    const unsigned short* attT = att_ws + ((size_t)t * N_ROWS + row0) * D_MODEL;
    const unsigned short* rmT  = rm_ws  + ((size_t)t * N_ROWS + row0) * D_MODEL;

    // --- stage 0: edge gates + emb tile -> S1 ------------------------------
    for (int i = tid; i < 3 * TILE_M; i += NTHREADS) {
        int k = i / TILE_M, r = i - k * TILE_M;
        rwv[i] = (k < K) ? rwg[(size_t)relEdge[t][k] * N_ROWS + row0 + r] : 0.0f;
    }
    for (int i = tid; i < TILE_M * D_MODEL; i += NTHREADS) {
        int r = i / D_MODEL, c = i - r * D_MODEL;
        S1[i] = f2bf(tables[((size_t)t * N_ROWS + row0 + r) * D_MODEL + c]);
    }
    __syncthreads();

    // --- stage 1: qp = emb @ Wq^T + bq -> S2 -------------------------------
    for (int ch = wave; ch < D_MODEL / 16; ch += NWAVES) {
        int n0 = ch * 16;
        v8f acc[MSUB]; init_acc(acc, in_b[n0 + nlo]);
        gemm4_acc(S1, D_MODEL, D_MODEL, Wq, D_MODEL, 0, n0, lane, acc);
        store_acc_bf16(S2, D_MODEL, n0, lane, acc, false);
    }
    __syncthreads();

    // --- stage 2: per-key K-projection (-> S1) + attention scores ----------
    for (int k = 0; k < K; ++k) {
        const int other = relOther[t][k];
        for (int i = tid; i < TILE_M * D_MODEL; i += NTHREADS) {
            int r = i / D_MODEL, c = i - r * D_MODEL;
            float x = tables[((size_t)other * N_ROWS + row0 + r) * D_MODEL + c];
            S3[i] = f2bf(x * rwv[k * TILE_M + r]);
        }
        __syncthreads();
        for (int ch = wave; ch < D_MODEL / 16; ch += NWAVES) {
            int n0 = ch * 16;
            v8f acc[MSUB]; init_acc(acc, in_b[D_MODEL + n0 + nlo]);
            gemm4_acc(S3, D_MODEL, D_MODEL, Wk, D_MODEL, 0, n0, lane, acc);
            store_acc_bf16(S1, D_MODEL, n0, lane, acc, false);   // kp
        }
        __syncthreads();
        for (int i = tid; i < TILE_M * N_HEADS; i += NTHREADS) {
            int r = i / N_HEADS, h = i - r * N_HEADS;
            int base = r * D_MODEL + h * HEAD_DIM;
            float s = 0.0f;
            for (int c = 0; c < HEAD_DIM; ++c)
                s += bf2f(S2[base + c]) * bf2f(S1[base + c]);    // qp . kp
            scores[(k * TILE_M + r) * N_HEADS + h] = s * 0.125f; // 1/sqrt(64)
        }
        __syncthreads();
    }

    // --- stage 3: softmax over K; oattn(bf16) = 0 -> S2 (qp dead) ----------
    for (int i = tid; i < TILE_M * N_HEADS; i += NTHREADS) {
        int r = i / N_HEADS, h = i - r * N_HEADS;
        float m = -3.0e38f;
        for (int k = 0; k < K; ++k)
            m = fmaxf(m, scores[(k * TILE_M + r) * N_HEADS + h]);
        float ssum = 0.0f;
        for (int k = 0; k < K; ++k) {
            float e = __expf(scores[(k * TILE_M + r) * N_HEADS + h] - m);
            attnw[(k * TILE_M + r) * N_HEADS + h] = e;
            ssum += e;
        }
        float inv = 1.0f / ssum;
        for (int k = 0; k < K; ++k)
            attnw[(k * TILE_M + r) * N_HEADS + h] *= inv;
    }
    __syncthreads();   // scores fully read before S2 reuse
    for (int i = tid; i < TILE_M * D_MODEL; i += NTHREADS) S2[i] = 0;
    __syncthreads();

    // --- stage 4: per-key V-projection, attention-weighted accumulate ------
    for (int k = 0; k < K; ++k) {
        const int other = relOther[t][k];
        for (int i = tid; i < TILE_M * D_MODEL; i += NTHREADS) {
            int r = i / D_MODEL, c = i - r * D_MODEL;
            float x = tables[((size_t)other * N_ROWS + row0 + r) * D_MODEL + c];
            S3[i] = f2bf(x * rwv[k * TILE_M + r]);
        }
        __syncthreads();
        for (int ch = wave; ch < D_MODEL / 16; ch += NWAVES) {
            int n0 = ch * 16;
            int head = n0 >> 6;
            v8f acc[MSUB]; init_acc(acc, in_b[2 * D_MODEL + n0 + nlo]);
            gemm4_acc(S3, D_MODEL, D_MODEL, Wv, D_MODEL, 0, n0, lane, acc);
            #pragma unroll
            for (int s = 0; s < MSUB; ++s)
                #pragma unroll
                for (int v = 0; v < 8; ++v) {
                    int m = s * 16 + hh * 8 + v;
                    size_t idx = (size_t)m * D_MODEL + n0 + nlo;
                    float w = attnw[(k * TILE_M + m) * N_HEADS + head];
                    S2[idx] = f2bf(bf2f(S2[idx]) + acc[s][v] * w);
                }
        }
        __syncthreads();
    }

    // --- stage 5: attended = oattn @ Wo^T + bo -> workspace (bf16) ---------
    for (int ch = wave; ch < D_MODEL / 16; ch += NWAVES) {
        int n0 = ch * 16;
        v8f acc[MSUB]; init_acc(acc, out_b[n0 + nlo]);
        gemm4_acc(S2, D_MODEL, D_MODEL, Wo, D_MODEL, 0, n0, lane, acc);
        unsigned short* dst = att_ws + ((size_t)t * N_ROWS + row0) * D_MODEL;
        #pragma unroll
        for (int s = 0; s < MSUB; ++s)
            #pragma unroll
            for (int v = 0; v < 8; ++v)
                dst[(size_t)(s * 16 + hh * 8 + v) * D_MODEL + n0 + nlo] =
                    f2bf(acc[s][v]);
    }

    // --- stage 6: rel_mean -> workspace (bf16); rebuild emb -> S1 ----------
    for (int i = tid; i < TILE_M * D_MODEL; i += NTHREADS) {
        int r = i / D_MODEL, c = i - r * D_MODEL;
        float s = 0.0f;
        for (int k = 0; k < K; ++k)
            s += rels[((size_t)relEdge[t][k] * N_ROWS + row0 + r) * D_MODEL + c];
        rm_ws[((size_t)t * N_ROWS + row0 + r) * D_MODEL + c] =
            f2bf(s * (1.0f / (float)K));
        S1[i] = f2bf(tables[((size_t)t * N_ROWS + row0 + r) * D_MODEL + c]);
    }
    __syncthreads();

    // --- stage 7: hidden = relu(combined @ W1^T + b1) -> hid (S2+S3) -------
    // combined blocks: [ S1(emb,LDS) | att_ws(global,L2-hot) | rm_ws ]
    for (int ch = wave; ch < (2 * D_MODEL) / 16; ch += NWAVES) {
        int n0 = ch * 16;
        v8f acc[MSUB]; init_acc(acc, b1[n0 + nlo]);
        gemm4_acc(S1,   D_MODEL, D_MODEL, W1b, 3 * D_MODEL, 0,           n0, lane, acc);
        gemm4_acc(attT, D_MODEL, D_MODEL, W1b, 3 * D_MODEL, D_MODEL,     n0, lane, acc);
        gemm4_acc(rmT,  D_MODEL, D_MODEL, W1b, 3 * D_MODEL, 2 * D_MODEL, n0, lane, acc);
        store_acc_bf16(hid_bf, 2 * D_MODEL, n0, lane, acc, true);
    }
    __syncthreads();

    // --- stage 8: fused / gate / blend -> global output --------------------
    for (int ch = wave; ch < D_MODEL / 16; ch += NWAVES) {
        int n0 = ch * 16;
        v8f f[MSUB]; init_acc(f, b2[n0 + nlo]);
        gemm4_acc(hid_bf, 2 * D_MODEL, 2 * D_MODEL, W2b, 2 * D_MODEL, 0, n0, lane, f);
        v8f g[MSUB]; init_acc(g, gb[n0 + nlo]);
        gemm4_acc(S1,   D_MODEL, D_MODEL, Wgb, 3 * D_MODEL, 0,           n0, lane, g);
        gemm4_acc(attT, D_MODEL, D_MODEL, Wgb, 3 * D_MODEL, D_MODEL,     n0, lane, g);
        gemm4_acc(rmT,  D_MODEL, D_MODEL, Wgb, 3 * D_MODEL, 2 * D_MODEL, n0, lane, g);
        #pragma unroll
        for (int s = 0; s < MSUB; ++s)
            #pragma unroll
            for (int v = 0; v < 8; ++v) {
                int m = s * 16 + hh * 8 + v;
                float gate = 1.0f / (1.0f + __expf(-g[s][v]));
                size_t idx = ((size_t)t * N_ROWS + row0 + m) * D_MODEL + n0 + nlo;
                float e = tables[idx];
                out[idx] = f[s][v] * gate + e * (1.0f - gate);
            }
    }
}

// ---------------------------------------------------------------------------
// Host entry
// ---------------------------------------------------------------------------
extern "C" void kernel_launch(void* const* d_in, const int* in_sizes, int n_in,
                              void* d_out, int out_size, void* d_ws, size_t ws_size,
                              hipStream_t stream) {
    const float* tables     = (const float*)d_in[0];
    const float* rels       = (const float*)d_in[1];
    const float* in_proj_w  = (const float*)d_in[2];
    const float* in_proj_b  = (const float*)d_in[3];
    const float* out_proj_w = (const float*)d_in[4];
    const float* out_proj_b = (const float*)d_in[5];
    const float* rel_w      = (const float*)d_in[6];
    const float* rel_b      = (const float*)d_in[7];
    const float* W1         = (const float*)d_in[8];
    const float* b1         = (const float*)d_in[9];
    const float* W2         = (const float*)d_in[10];
    const float* b2         = (const float*)d_in[11];
    const float* gate_w     = (const float*)d_in[12];
    const float* gate_b     = (const float*)d_in[13];
    float* out              = (float*)d_out;

    const size_t S_DD = (size_t)D_MODEL * D_MODEL;          // 589824
    const size_t S_W1 = (size_t)2 * D_MODEL * 3 * D_MODEL;  // 3538944
    const size_t S_W2 = (size_t)D_MODEL * 2 * D_MODEL;      // 1179648
    const size_t S_WG = (size_t)D_MODEL * 3 * D_MODEL;      // 1769472
    const size_t S_ND = (size_t)N_TABLES * N_ROWS * D_MODEL;// 50331648

    char* ws = (char*)d_ws;
    size_t off = 0;
    unsigned short* Wq  = (unsigned short*)(ws + off); off += S_DD * 2;
    unsigned short* Wk  = (unsigned short*)(ws + off); off += S_DD * 2;
    unsigned short* Wv  = (unsigned short*)(ws + off); off += S_DD * 2;
    unsigned short* Wo  = (unsigned short*)(ws + off); off += S_DD * 2;
    unsigned short* W1b = (unsigned short*)(ws + off); off += S_W1 * 2;
    unsigned short* W2b = (unsigned short*)(ws + off); off += S_W2 * 2;
    unsigned short* Wgb = (unsigned short*)(ws + off); off += S_WG * 2;
    float* rwg = (float*)(ws + off);                   off += (size_t)N_EDGES * N_ROWS * 4;
    unsigned short* att_ws = (unsigned short*)(ws + off); off += S_ND * 2;
    unsigned short* rm_ws  = (unsigned short*)(ws + off); off += S_ND * 2;
    // total workspace: ~18 MB weights + ~200 MB bf16 intermediates

    auto cvt = [&](const float* s, unsigned short* d, size_t n) {
        cvt_bf16_kernel<<<(unsigned)((n + 255) / 256), 256, 0, stream>>>(s, d, (int)n);
    };
    cvt(in_proj_w,              Wq,  S_DD);
    cvt(in_proj_w + S_DD,       Wk,  S_DD);
    cvt(in_proj_w + 2 * S_DD,   Wv,  S_DD);
    cvt(out_proj_w,             Wo,  S_DD);
    cvt(W1,                     W1b, S_W1);
    cvt(W2,                     W2b, S_W2);
    cvt(gate_w,                 Wgb, S_WG);

    rw_kernel<<<(N_EDGES * N_ROWS) / 8, 256, 0, stream>>>(rels, rel_w, rel_b, rwg);

    dim3 grid(N_ROWS / TILE_M, N_TABLES);
    graph_encoder_main<<<grid, NTHREADS, SMEM_BYTES, stream>>>(
        tables, rels, Wq, Wk, Wv, Wo, W1b, W2b, Wgb, rwg, att_ws, rm_ws,
        in_proj_b, out_proj_b, b1, b2, gate_b, out);
}
